// GCN_71201967833969
// MI455X (gfx1250) — compile-verified, matrix operand order checked by reading, not run on hardware
//
#include <hip/hip_runtime.h>
#include <hip/hip_bf16.h>

typedef __attribute__((ext_vector_type(2)))  float  v2f;
typedef __attribute__((ext_vector_type(8)))  float  v8f;
typedef __attribute__((ext_vector_type(16))) __bf16 v16bf;

#if defined(__has_builtin)
#if __has_builtin(__builtin_amdgcn_wmma_f32_16x16x4_f32)
#define USE_F32_WMMA 1
#else
#define USE_F32_WMMA 0
#endif
#else
#define USE_F32_WMMA 0
#endif

// ---------------------------------------------------------------------------
// Degree accumulation: deg[dst] += w over E edges, +1.0 self-loop per node.
// ---------------------------------------------------------------------------
__global__ void deg_kernel(const long long* __restrict__ dst,
                           const float* __restrict__ w,
                           float* __restrict__ deg, int nE, int nN) {
    int i = blockIdx.x * blockDim.x + threadIdx.x;
    if (i < nE) {
        atomicAdd(&deg[(int)dst[i]], w[i]);
    } else if (i < nE + nN) {
        atomicAdd(&deg[i - nE], 1.0f);
    }
}

__global__ void rsqrt_kernel(float* __restrict__ deg, int n) {
    int i = blockIdx.x * blockDim.x + threadIdx.x;
    if (i < n) {
        float d = deg[i];
        deg[i] = (d > 0.0f) ? rsqrtf(d) : 0.0f;
    }
}

// ---------------------------------------------------------------------------
// WMMA GEMM: H[nrows x 128] = X[nrows x K] @ W[K x 128]
// 256 threads = 8 waves; each wave owns a 16-row strip and all 128 columns
// (8 accumulator tiles of 16x16). nrows is a multiple of 16, so the
// out-of-range guard is wave-uniform -> EXEC stays all-1s for WMMA.
// ---------------------------------------------------------------------------
template <int K>
__global__ void gemm_wmma(const float* __restrict__ X,
                          const float* __restrict__ W,
                          float* __restrict__ H, int nrows) {
    const int lane = threadIdx.x & 31;
    const int wave = threadIdx.x >> 5;
    const int r0 = (blockIdx.x * 8 + wave) * 16;
    if (r0 >= nrows) return;                 // wave-uniform exit

    const int m  = lane & 15;                // row within tile (A), col (B/C/D)
    const int hb = (lane >> 4) << 3;         // 0 or 8: lane-half offset

    v8f acc[8] = {};
    const float* xrow = X + (size_t)(r0 + m) * K;

#if USE_F32_WMMA
    // fp32 16x16x4: A VGPR0/1 = K pair; lanes 0-15 hold K=0,1; lanes16-31 K=2,3
    const int kk = (lane >> 4) << 1;         // 0 or 2
    for (int k = 0; k < K; k += 4) {
        v2f a = *(const v2f*)(xrow + k + kk);            // 8B aligned (kk even)
#pragma unroll
        for (int t = 0; t < 8; ++t) {
            const int n = (t << 4) + m;
            v2f b;
            b.x = W[(size_t)(k + kk    ) * 128 + n];
            b.y = W[(size_t)(k + kk + 1) * 128 + n];
            acc[t] = __builtin_amdgcn_wmma_f32_16x16x4_f32(
                false, a, false, b, (short)0, acc[t], false, false);
        }
    }
#else
    // bf16 16x16x32 fallback (codegen-confirmed builtin), fp32 accumulate.
    for (int k = 0; k < K; k += 32) {
        v16bf a;
#pragma unroll
        for (int j = 0; j < 16; ++j) {
            const int v = j >> 1, bit = j & 1;
            const int kj = ((v < 4) ? (v << 1) : (16 + ((v - 4) << 1))) + hb + bit;
            a[j] = (__bf16)xrow[k + kj];
        }
#pragma unroll
        for (int t = 0; t < 8; ++t) {
            const int n = (t << 4) + m;
            v16bf b;
#pragma unroll
            for (int j = 0; j < 16; ++j) {
                const int v = j >> 1, bit = j & 1;
                const int kj = ((v < 4) ? (v << 1) : (16 + ((v - 4) << 1))) + hb + bit;
                b[j] = (__bf16)W[(size_t)(k + kj) * 128 + n];
            }
            acc[t] = __builtin_amdgcn_wmma_f32_16x16x32_bf16(
                false, a, false, b, (short)0, acc[t], false, false);
        }
    }
#endif

    // C/D layout: VGPR i, lanes 0-15 -> M=i ; lanes 16-31 -> M=8+i
#pragma unroll
    for (int t = 0; t < 8; ++t) {
        const int n = (t << 4) + m;
#pragma unroll
        for (int i = 0; i < 8; ++i) {
            H[(size_t)(r0 + hb + i) * 128 + n] = acc[t][i];
        }
    }
}

// ---------------------------------------------------------------------------
// Edge scatter-add: one wave32 per (edge | self-loop).
// out[dst] += dinv[src]*w*dinv[dst] * H[src]; lane c covers cols c+{0,32,64,96}
// -> fully coalesced atomic_add_f32 bursts that stay resident in L2 (51 MB).
// ---------------------------------------------------------------------------
__global__ void scatter_edges(const long long* __restrict__ srcI,
                              const long long* __restrict__ dstI,
                              const float* __restrict__ w,
                              const float* __restrict__ dinv,
                              const float* __restrict__ H,
                              float* __restrict__ out, int nE, int nN) {
    const int e = blockIdx.x * (blockDim.x >> 5) + (threadIdx.x >> 5);
    const int lane = threadIdx.x & 31;
    if (e >= nE + nN) return;

    int s, d;
    float ww;
    if (e < nE) {
        s = (int)srcI[e];
        d = (int)dstI[e];
        ww = w[e];
    } else {
        s = d = e - nE;
        ww = 1.0f;
    }
    const float norm = dinv[s] * ww * dinv[d];
    const float* hs = H + (size_t)s * 128;
    float* od = out + (size_t)d * 128;
#pragma unroll
    for (int c = lane; c < 128; c += 32) {
        atomicAdd(od + c, norm * hs[c]);
    }
}

// ---------------------------------------------------------------------------
// Elementwise bias (+ optional ReLU), in place. cols == 128 -> col = i & 127.
// ---------------------------------------------------------------------------
template <int RELU>
__global__ void bias_act(float* __restrict__ a, const float* __restrict__ b,
                         size_t total) {
    size_t i = (size_t)blockIdx.x * blockDim.x + threadIdx.x;
    if (i < total) {
        float v = a[i] + b[i & 127];
        if (RELU) v = fmaxf(v, 0.0f);
        a[i] = v;
    }
}

static inline int cdiv(long long a, long long b) { return (int)((a + b - 1) / b); }

extern "C" void kernel_launch(void* const* d_in, const int* in_sizes, int n_in,
                              void* d_out, int out_size, void* d_ws, size_t ws_size,
                              hipStream_t stream) {
    const float*     x  = (const float*)d_in[0];     // [N,256]
    const long long* ei = (const long long*)d_in[1]; // [2,E] int64
    const float*     ew = (const float*)d_in[2];     // [E]
    const float*     W1 = (const float*)d_in[3];     // [256,128]
    const float*     b1 = (const float*)d_in[4];     // [128]
    const float*     W2 = (const float*)d_in[5];     // [128,128]
    const float*     b2 = (const float*)d_in[6];     // [128]
    float*           out = (float*)d_out;            // [N,128]

    const int N = in_sizes[0] / 256;
    const int E = in_sizes[2];
    const long long* srcI = ei;
    const long long* dstI = ei + E;

    const size_t feat = (size_t)N * 128;
    float* bufA = (float*)d_ws;          // GEMM outputs (overwritten each use)
    float* bufB = bufA + feat;           // layer-1 aggregation / hidden h
    float* dinv = bufB + feat;           // degrees -> dinv, [N]

    // Zero accumulation targets every call (harness does not re-poison).
    hipMemsetAsync(dinv, 0, (size_t)N * sizeof(float), stream);
    hipMemsetAsync(bufB, 0, feat * sizeof(float), stream);
    hipMemsetAsync(out,  0, feat * sizeof(float), stream);

    // Normalization coefficients.
    deg_kernel<<<cdiv(E + N, 256), 256, 0, stream>>>(dstI, ew, dinv, E, N);
    rsqrt_kernel<<<cdiv(N, 256), 256, 0, stream>>>(dinv, N);

    const int gemm_blocks = cdiv(N, 128);        // 8 waves x 16 rows per block
    const int scat_blocks = cdiv(E + N, 8);      // 8 edge-waves per block
    const int ew_blocks   = cdiv((long long)feat, 256);

    // Layer 1: h = relu(scatter(norm * (x@W1)) + b1)
    gemm_wmma<256><<<gemm_blocks, 256, 0, stream>>>(x, W1, bufA, N);
    scatter_edges<<<scat_blocks, 256, 0, stream>>>(srcI, dstI, ew, dinv, bufA, bufB, E, N);
    bias_act<1><<<ew_blocks, 256, 0, stream>>>(bufB, b1, feat);

    // Layer 2: out = scatter(norm * (h@W2)) + b2
    gemm_wmma<128><<<gemm_blocks, 256, 0, stream>>>(bufB, W2, bufA, N);
    scatter_edges<<<scat_blocks, 256, 0, stream>>>(srcI, dstI, ew, dinv, bufA, out, E, N);
    bias_act<0><<<ew_blocks, 256, 0, stream>>>(out, b2, feat);
}
